// AdaptiveExpertSystem_43439299232051
// MI455X (gfx1250) — matrix operand order repaired and checked by reading, not compiled, not run on hardware
//
#include <hip/hip_runtime.h>
#include <math.h>

typedef __attribute__((ext_vector_type(16))) _Float16 v16h;
typedef __attribute__((ext_vector_type(8)))  _Float16 v8h;
typedef __attribute__((ext_vector_type(4)))  _Float16 v4h;
typedef __attribute__((ext_vector_type(8)))  float    v8f;
typedef __attribute__((ext_vector_type(4)))  float    v4f;

#define T_TOK 8192
#define HID   1024
#define INTERD 4096
#define NEXP  8
#define BM 128
#define BN 128
#define BK 32
#define LDP 40   // LDS row stride in halfs (80B: 16B-aligned rows, conflict-free)

union HFrag { v16h v; v8h h[2]; };

__device__ __forceinline__ uint32_t to_lds(const void* p) {
  return (uint32_t)(uintptr_t)p;   // generic shared ptr: low 32 bits = LDS offset
}
__device__ __forceinline__ void async_b128(uint32_t lds, const void* g) {
  asm volatile("global_load_async_to_lds_b128 %0, %1, off"
               :: "v"(lds), "v"(g) : "memory");
}
__device__ __forceinline__ void async_b128_off16(uint32_t lds, const void* g) {
  // INST_OFFSET applies to both LDS and global address (ISA async-LDS pseudocode)
  asm volatile("global_load_async_to_lds_b128 %0, %1, off offset:16"
               :: "v"(lds), "v"(g) : "memory");
}
__device__ __forceinline__ void wait_async0() {
  asm volatile("s_wait_asynccnt 0x0" ::: "memory");
}

__device__ __forceinline__ v16h load_afrag(const _Float16* p) { // &As[r0+fm][fh*8]
  HFrag f; f.h[0] = *(const v8h*)p; f.h[1] = *(const v8h*)(p + 16); return f.v;
}
__device__ __forceinline__ v16h load_bfrag(const _Float16* p) { // &Bs[c0+fm][fh*8]
  HFrag f; f.h[0] = *(const v8h*)p; f.h[1] = *(const v8h*)(p + 16); return f.v;
}
#define WMMA(A, B, C) __builtin_amdgcn_wmma_f32_16x16x32_f16(false, A, false, B, (short)0, C, false, false)

// ---------------- init: zero output + per-expert counters ----------------
__global__ void zero_kernel(float* __restrict__ out, int n, int* __restrict__ counts) {
  int i = blockIdx.x * blockDim.x + threadIdx.x;
  if (i < n) out[i] = 0.0f;
  if (blockIdx.x == 0 && threadIdx.x < NEXP) counts[threadIdx.x] = 0;
}

// ---------------- fp32 -> f16 weight staging ----------------
__global__ void cvt_kernel(const float* __restrict__ s, _Float16* __restrict__ d, int n) {
  int i = (blockIdx.x * blockDim.x + threadIdx.x) * 4;
  if (i < n) {
    v4f f = *(const v4f*)(s + i);
    v4h h; h[0]=(_Float16)f[0]; h[1]=(_Float16)f[1]; h[2]=(_Float16)f[2]; h[3]=(_Float16)f[3];
    *(v4h*)(d + i) = h;
  }
}

// ---------------- router: LN + logits + top2 softmax + gather lists ------
__global__ __launch_bounds__(256) void router_kernel(
    const float* __restrict__ x, const float* __restrict__ lns,
    const float* __restrict__ lnb, const float* __restrict__ rw,
    const float* __restrict__ rb,
    _Float16* __restrict__ xh, int* __restrict__ counts,
    int* __restrict__ rowlist, float* __restrict__ gatelist) {
  const int t   = blockIdx.x;
  const int tid = threadIdx.x;
  const float* xr = x + (size_t)t * HID;

  float v[4], s = 0.f, sq = 0.f;
#pragma unroll
  for (int i = 0; i < 4; ++i) {
    v[i] = xr[tid * 4 + i];
    s += v[i]; sq += v[i] * v[i];
    xh[(size_t)t * HID + tid * 4 + i] = (_Float16)v[i];
  }

  __shared__ float red[512];
  red[tid] = s; red[256 + tid] = sq;
  __syncthreads();
  for (int st = 128; st > 0; st >>= 1) {
    if (tid < st) { red[tid] += red[tid + st]; red[256 + tid] += red[256 + tid + st]; }
    __syncthreads();
  }
  __shared__ float mu_s, rs_s;
  if (tid == 0) {
    float mu = red[0] * (1.0f / HID);
    float var = red[256] * (1.0f / HID) - mu * mu;
    mu_s = mu; rs_s = rsqrtf(var + 1e-5f);
  }
  __syncthreads();
  const float mu = mu_s, rstd = rs_s;

  float part[NEXP];
#pragma unroll
  for (int e = 0; e < NEXP; ++e) part[e] = 0.f;
#pragma unroll
  for (int i = 0; i < 4; ++i) {
    int idx = tid * 4 + i;
    float xn = (v[i] - mu) * rstd * lns[idx] + lnb[idx];
#pragma unroll
    for (int e = 0; e < NEXP; ++e) part[e] += xn * rw[e * HID + idx];
  }

  __shared__ float red8[256 * NEXP];
#pragma unroll
  for (int e = 0; e < NEXP; ++e) red8[tid * NEXP + e] = part[e];
  __syncthreads();
  for (int st = 128; st > 0; st >>= 1) {
    if (tid < st)
#pragma unroll
      for (int e = 0; e < NEXP; ++e)
        red8[tid * NEXP + e] += red8[(tid + st) * NEXP + e];
    __syncthreads();
  }

  if (tid == 0) {
    float lg[NEXP];
#pragma unroll
    for (int e = 0; e < NEXP; ++e) lg[e] = red8[e] + rb[e];
    int i0 = 0; float b0v = lg[0];
#pragma unroll
    for (int e = 1; e < NEXP; ++e) if (lg[e] > b0v) { b0v = lg[e]; i0 = e; }
    int i1 = (i0 == 0) ? 1 : 0; float b1v = lg[i1];
#pragma unroll
    for (int e = 0; e < NEXP; ++e)
      if (e != i0 && lg[e] > b1v) { b1v = lg[e]; i1 = e; }
    float g0 = 1.0f / (1.0f + expf(b1v - b0v));
    float g1 = 1.0f - g0;
    int s0 = atomicAdd(&counts[i0], 1);
    rowlist[i0 * T_TOK + s0] = t; gatelist[i0 * T_TOK + s0] = g0;
    int s1 = atomicAdd(&counts[i1], 1);
    rowlist[i1 * T_TOK + s1] = t; gatelist[i1 * T_TOK + s1] = g1;
  }
}

// ---------------- up GEMM: h = gelu(x @ up_w^T + up_b), gathered rows ----
__global__ __launch_bounds__(256) void up_gemm_kernel(
    const _Float16* __restrict__ Xh,     // [T][HID] f16
    const int* __restrict__ rowlist,
    const int* __restrict__ countp,
    const _Float16* __restrict__ Wh,     // [INTERD][HID] f16 (staged)
    const float* __restrict__ bias,      // [INTERD]
    _Float16* __restrict__ Hbuf) {       // [slot][INTERD] f16
  const int count = *countp;
  const int rowBase = blockIdx.y * BM;
  if (rowBase >= count) return;
  const int colBase = blockIdx.x * BN;
  const int tid = threadIdx.x;

  __shared__ __align__(16) _Float16 As[2][BM][LDP];
  __shared__ __align__(16) _Float16 Bs[2][BN][LDP];
  __shared__ int toks[BM];

  if (tid < BM) {
    int slot = rowBase + tid;
    if (slot >= count) slot = count - 1;
    toks[tid] = rowlist[slot];
  }
  __syncthreads();

  const int ar = tid >> 1;            // load row 0..127 (A and B tiles)
  const int ak = (tid & 1) * 16;      // halfs offset: each thread covers 16 halfs (2x16B)

  const int lane = tid & 31;
  const int w    = tid >> 5;
  const int wr   = w >> 1;            // 0..3 : wave row (32 rows)
  const int wc   = w & 1;             // 0..1 : wave col (64 cols)
  const int fm   = lane & 15;
  const int fh   = lane >> 4;

  const uint32_t lA0 = to_lds(&As[0][ar][ak]);
  const uint32_t lA1 = to_lds(&As[1][ar][ak]);
  const uint32_t lB0 = to_lds(&Bs[0][ar][ak]);
  const uint32_t lB1 = to_lds(&Bs[1][ar][ak]);
  const _Float16* fetchA = Xh + (size_t)toks[ar] * HID + ak;
  const _Float16* fetchB = Wh + (size_t)(colBase + ar) * HID + ak;

  v8f acc[2][4] = {};

  auto issue = [&](int buf) {
    uint32_t la = buf ? lA1 : lA0;
    uint32_t lb = buf ? lB1 : lB0;
    async_b128(la, fetchA); async_b128_off16(la, fetchA);
    async_b128(lb, fetchB); async_b128_off16(lb, fetchB);
    fetchA += BK; fetchB += BK;
  };
  auto compute = [&](int buf) {
    v16h a0 = load_afrag(&As[buf][wr * 32 + fm][fh * 8]);
    v16h a1 = load_afrag(&As[buf][wr * 32 + 16 + fm][fh * 8]);
    v16h b[4];
#pragma unroll
    for (int j = 0; j < 4; ++j)
      b[j] = load_bfrag(&Bs[buf][wc * 64 + j * 16 + fm][fh * 8]);
#pragma unroll
    for (int j = 0; j < 4; ++j) {
      acc[0][j] = WMMA(a0, b[j], acc[0][j]);
      acc[1][j] = WMMA(a1, b[j], acc[1][j]);
    }
  };

  issue(0);
  wait_async0();
  __syncthreads();
  for (int kt = 0; kt < HID; kt += 2 * BK) {
    issue(1);
    compute(0);
    wait_async0();
    __syncthreads();
    if (kt + 2 * BK < HID) issue(0);
    compute(1);
    wait_async0();
    __syncthreads();
  }

  const int nl = lane & 15;
  const int mg = fh * 8;
#pragma unroll
  for (int i = 0; i < 2; ++i) {
#pragma unroll
    for (int v = 0; v < 8; ++v) {
      int r = wr * 32 + i * 16 + mg + v;
      int slot = rowBase + r;
      if (slot < count) {
#pragma unroll
        for (int j = 0; j < 4; ++j) {
          int c = colBase + wc * 64 + j * 16 + nl;
          float t0 = acc[i][j][v] + bias[c];
          t0 = 0.5f * t0 * (1.0f + erff(t0 * 0.70710678118654752f));
          Hbuf[(size_t)slot * INTERD + c] = (_Float16)t0;
        }
      }
    }
  }
}

// ---------------- down GEMM: out[token] += gate*(h @ down_w^T + down_b) --
__global__ __launch_bounds__(256) void down_gemm_kernel(
    const _Float16* __restrict__ Hbuf,   // [slot][INTERD] f16
    const int* __restrict__ rowlist,
    const float* __restrict__ gatelist,
    const int* __restrict__ countp,
    const _Float16* __restrict__ Wh,     // [HID][INTERD] f16 (staged)
    const float* __restrict__ bias,      // [HID]
    float* __restrict__ out) {           // [T][HID]
  const int count = *countp;
  const int rowBase = blockIdx.y * BM;
  if (rowBase >= count) return;
  const int colBase = blockIdx.x * BN;
  const int tid = threadIdx.x;

  __shared__ __align__(16) _Float16 As[2][BM][LDP];
  __shared__ __align__(16) _Float16 Bs[2][BN][LDP];
  __shared__ int   toks[BM];
  __shared__ float gts[BM];

  if (tid < BM) {
    int slot = rowBase + tid;
    int cs = (slot < count) ? slot : (count - 1);
    toks[tid] = rowlist[cs];
    gts[tid]  = gatelist[cs];
  }
  __syncthreads();

  const int ar = tid >> 1;
  const int ak = (tid & 1) * 16;

  const int lane = tid & 31;
  const int w    = tid >> 5;
  const int wr   = w >> 1;
  const int wc   = w & 1;
  const int fm   = lane & 15;
  const int fh   = lane >> 4;

  const uint32_t lA0 = to_lds(&As[0][ar][ak]);
  const uint32_t lA1 = to_lds(&As[1][ar][ak]);
  const uint32_t lB0 = to_lds(&Bs[0][ar][ak]);
  const uint32_t lB1 = to_lds(&Bs[1][ar][ak]);
  int arow = rowBase + ar;
  if (arow >= count) arow = count - 1;
  const _Float16* fetchA = Hbuf + (size_t)arow * INTERD + ak;
  const _Float16* fetchB = Wh + (size_t)(colBase + ar) * INTERD + ak;

  v8f acc[2][4] = {};

  auto issue = [&](int buf) {
    uint32_t la = buf ? lA1 : lA0;
    uint32_t lb = buf ? lB1 : lB0;
    async_b128(la, fetchA); async_b128_off16(la, fetchA);
    async_b128(lb, fetchB); async_b128_off16(lb, fetchB);
    fetchA += BK; fetchB += BK;
  };
  auto compute = [&](int buf) {
    v16h a0 = load_afrag(&As[buf][wr * 32 + fm][fh * 8]);
    v16h a1 = load_afrag(&As[buf][wr * 32 + 16 + fm][fh * 8]);
    v16h b[4];
#pragma unroll
    for (int j = 0; j < 4; ++j)
      b[j] = load_bfrag(&Bs[buf][wc * 64 + j * 16 + fm][fh * 8]);
#pragma unroll
    for (int j = 0; j < 4; ++j) {
      acc[0][j] = WMMA(a0, b[j], acc[0][j]);
      acc[1][j] = WMMA(a1, b[j], acc[1][j]);
    }
  };

  issue(0);
  wait_async0();
  __syncthreads();
  for (int kt = 0; kt < INTERD; kt += 2 * BK) {
    issue(1);
    compute(0);
    wait_async0();
    __syncthreads();
    if (kt + 2 * BK < INTERD) issue(0);
    compute(1);
    wait_async0();
    __syncthreads();
  }

  const int nl = lane & 15;
  const int mg = fh * 8;
#pragma unroll
  for (int i = 0; i < 2; ++i) {
#pragma unroll
    for (int v = 0; v < 8; ++v) {
      int r = wr * 32 + i * 16 + mg + v;
      int slot = rowBase + r;
      if (slot < count) {
        int tok = toks[r]; float g = gts[r];
#pragma unroll
        for (int j = 0; j < 4; ++j) {
          int c = colBase + wc * 64 + j * 16 + nl;
          out[(size_t)tok * HID + c] += g * (acc[i][j][v] + bias[c]);
        }
      }
    }
  }
}

// ---------------- host launcher ----------------
extern "C" void kernel_launch(void* const* d_in, const int* in_sizes, int n_in,
                              void* d_out, int out_size, void* d_ws, size_t ws_size,
                              hipStream_t stream) {
  (void)in_sizes; (void)n_in; (void)out_size; (void)ws_size;
  const float* x   = (const float*)d_in[0];
  const float* lns = (const float*)d_in[1];
  const float* lnb = (const float*)d_in[2];
  const float* rw  = (const float*)d_in[3];
  const float* rb  = (const float*)d_in[4];
  const float* upw = (const float*)d_in[5];
  const float* upb = (const float*)d_in[6];
  const float* dww = (const float*)d_in[7];
  const float* dwb = (const float*)d_in[8];
  float* out = (float*)d_out;

  char* ws = (char*)d_ws;
  size_t off = 0;
  auto take = [&](size_t bytes) -> void* {
    void* p = ws + off;
    off = (off + bytes + 255) & ~(size_t)255;
    return p;
  };
  _Float16* xh       = (_Float16*)take((size_t)T_TOK * HID * 2);       // 16 MB
  int*      counts   = (int*)take(NEXP * sizeof(int));
  int*      rowlist  = (int*)take((size_t)NEXP * T_TOK * 4);           // 256 KB
  float*    gatelist = (float*)take((size_t)NEXP * T_TOK * 4);         // 256 KB
  _Float16* Hbuf     = (_Float16*)take((size_t)T_TOK * INTERD * 2);    // 64 MB
  _Float16* WupH     = (_Float16*)take((size_t)INTERD * HID * 2);      // 8 MB
  _Float16* WdnH     = (_Float16*)take((size_t)HID * INTERD * 2);      // 8 MB

  const int nOut = T_TOK * HID;
  const int nW = INTERD * HID;
  zero_kernel<<<(nOut + 255) / 256, 256, 0, stream>>>(out, nOut, counts);
  router_kernel<<<T_TOK, 256, 0, stream>>>(x, lns, lnb, rw, rb,
                                           xh, counts, rowlist, gatelist);
  for (int e = 0; e < NEXP; ++e) {
    cvt_kernel<<<nW / 4 / 256, 256, 0, stream>>>(upw + (size_t)e * nW, WupH, nW);
    up_gemm_kernel<<<dim3(INTERD / BN, T_TOK / BM), 256, 0, stream>>>(
        xh, rowlist + (size_t)e * T_TOK, counts + e,
        WupH, upb + (size_t)e * INTERD, Hbuf);
    cvt_kernel<<<nW / 4 / 256, 256, 0, stream>>>(dww + (size_t)e * nW, WdnH, nW);
    down_gemm_kernel<<<dim3(HID / BN, T_TOK / BM), 256, 0, stream>>>(
        Hbuf, rowlist + (size_t)e * T_TOK, gatelist + (size_t)e * T_TOK,
        counts + e, WdnH, dwb + (size_t)e * HID, out);
  }
}
